// VQVAE_86818468922012
// MI455X (gfx1250) — matrix operand order repaired and checked by reading, not compiled, not run on hardware
//
#include <hip/hip_runtime.h>
#include <hip/hip_bf16.h>

typedef __attribute__((ext_vector_type(16))) __bf16 v16bf;
typedef __attribute__((ext_vector_type(8)))  float  v8f;

#define C_DIM            128
#define TILE_K           16     // codebook rows per k-tile
#define ROWS_PER_WAVE    16
#define WAVES_PER_BLOCK  8
#define ROWS_PER_BLOCK   128
#define LDS_STRIDE       144    // bf16 elems per staged row: 128 + pad, keeps 32B alignment

static __device__ __forceinline__ unsigned int pack_bf16(__bf16 lo, __bf16 hi)
{
    unsigned int l = __builtin_bit_cast(unsigned short, lo);
    unsigned int h = __builtin_bit_cast(unsigned short, hi);
    return (h << 16) | l;
}

// ---------------------------------------------------------------------------
// Kernel 0: exact fp32 ||codebook_k||^2 per row -> workspace (once).
// ---------------------------------------------------------------------------
__global__ __launch_bounds__(256) void vq_cbsq_kernel(
    const float* __restrict__ cb, float* __restrict__ cbsq, int K)
{
    const int k = blockIdx.x * blockDim.x + threadIdx.x;
    if (k >= K) return;
    const float* r = cb + (size_t)k * C_DIM;
    float s = 0.f;
    #pragma unroll
    for (int c = 0; c < C_DIM; c += 4) {
        float4 q = *(const float4*)(r + c);
        s += q.x * q.x + q.y * q.y + q.z * q.z + q.w * q.w;
    }
    cbsq[k] = s;
}

// ---------------------------------------------------------------------------
// Kernel 1: distances via bf16 hi/lo split WMMAs (near-fp32 accuracy),
// argmin, gather, loss partials. Double-buffered LDS codebook tiles.
// block = 256 threads (8 waves); each wave owns a 16-row z_e tile.
// ---------------------------------------------------------------------------
__global__ __launch_bounds__(256) void vq_argmin_kernel(
    const float* __restrict__ z_e, const float* __restrict__ cb,
    const float* __restrict__ cbsq_g,
    float* __restrict__ out_zq, float* __restrict__ out_idx,
    float* __restrict__ partial, int K)
{
    __shared__ __align__(32) __bf16 sBhi[2][TILE_K][LDS_STRIDE];
    __shared__ __align__(32) __bf16 sBlo[2][TILE_K][LDS_STRIDE];
    __shared__ float s_cbsq[2][TILE_K];
    __shared__ float s_loss[WAVES_PER_BLOCK];

    const int tid  = threadIdx.x;
    const int wave = tid >> 5;
    const int lane = tid & 31;
    const int half = lane >> 4;   // which 16-lane half
    const int l16  = lane & 15;
    const int n0   = blockIdx.x * ROWS_PER_BLOCK + wave * ROWS_PER_WAVE;

    // Cooperative stage of one 16x128 codebook tile -> LDS (bf16 hi/lo split),
    // packed pairwise so the LDS write is 2x ds_store_b128 per thread.
    auto stage = [&](int kt, int buf) {
        const int srow = tid >> 4;          // 0..15
        const int scol = (tid & 15) * 8;    // 0,8,...,120
        const float* cr = cb + (size_t)(kt * TILE_K + srow) * C_DIM + scol;
        float4 q0 = *(const float4*)(cr);
        float4 q1 = *(const float4*)(cr + 4);
        float xs[8] = {q0.x,q0.y,q0.z,q0.w, q1.x,q1.y,q1.z,q1.w};
        unsigned int hp[4], lp[4];
        #pragma unroll
        for (int p = 0; p < 4; ++p) {
            float x0 = xs[2*p], x1 = xs[2*p+1];
            __bf16 h0 = (__bf16)x0, h1 = (__bf16)x1;
            __bf16 g0 = (__bf16)(x0 - (float)h0), g1 = (__bf16)(x1 - (float)h1);
            hp[p] = pack_bf16(h0, h1);
            lp[p] = pack_bf16(g0, g1);
        }
        uint4 hv = {hp[0], hp[1], hp[2], hp[3]};
        uint4 lv = {lp[0], lp[1], lp[2], lp[3]};
        *(uint4*)&sBhi[buf][srow][scol] = hv;
        *(uint4*)&sBlo[buf][srow][scol] = lv;
        if (tid < TILE_K) s_cbsq[buf][tid] = cbsq_g[kt * TILE_K + tid];
    };

    // ---- Load and hi/lo-split the A (z_e) fragments once; resident across K loop.
    // A k order used here (same as B): lane holds K=16*half+e, e=0..15.
    v16bf a_hi[4], a_lo[4];
    {
        const float* zr = z_e + (size_t)(n0 + l16) * C_DIM;
        #pragma unroll
        for (int t = 0; t < 4; ++t) {
            const float* p = zr + 32 * t + 16 * half;
            float4 q0 = *(const float4*)(p);
            float4 q1 = *(const float4*)(p + 4);
            float4 q2 = *(const float4*)(p + 8);
            float4 q3 = *(const float4*)(p + 12);
            float xs[16] = {q0.x,q0.y,q0.z,q0.w, q1.x,q1.y,q1.z,q1.w,
                            q2.x,q2.y,q2.z,q2.w, q3.x,q3.y,q3.z,q3.w};
            #pragma unroll
            for (int e = 0; e < 16; ++e) {
                float x = xs[e];
                __bf16 h = (__bf16)x;
                a_hi[t][e] = h;
                a_lo[t][e] = (__bf16)(x - (float)h);
            }
        }
    }

    float best[8];
    int   bidx[8];
    #pragma unroll
    for (int j = 0; j < 8; ++j) { best[j] = 3.4e38f; bidx[j] = 0; }

    const int ktiles = K / TILE_K;

    stage(0, 0);
    __syncthreads();

    for (int kt = 0; kt < ktiles; ++kt) {
        const int buf = kt & 1;

        // ---- Hoist all B fragments for this tile into registers so the LDS
        // loads pipeline against the WMMA chain instead of blocking it.
        // B k order used here (same as A): per lane 16 contiguous K-values of column l16,
        // lanes>=16 take K+16 -> 16 contiguous bf16 from the staged codebook row.
        v16bf bh[4], bl[4];
        #pragma unroll
        for (int t = 0; t < 4; ++t) {
            bh[t] = *(const v16bf*)&sBhi[buf][l16][32 * t + 16 * half];
            bl[t] = *(const v16bf*)&sBlo[buf][l16][32 * t + 16 * half];
        }
        const float cbsq = s_cbsq[buf][l16];

        // ---- Stage the next tile into the other buffer (overlaps with WMMAs).
        if (kt + 1 < ktiles) stage(kt + 1, buf ^ 1);

        // ---- 12 bf16 WMMAs (hi*hi + hi*lo + lo*hi) -> near-fp32 16x16 cross tile.
        v8f acc = {0.f,0.f,0.f,0.f,0.f,0.f,0.f,0.f};
        #pragma unroll
        for (int t = 0; t < 4; ++t) {
            acc = __builtin_amdgcn_wmma_f32_16x16x32_bf16(false, a_hi[t], false, bh[t], (short)0, acc, false, false);
            acc = __builtin_amdgcn_wmma_f32_16x16x32_bf16(false, a_lo[t], false, bh[t], (short)0, acc, false, false);
            acc = __builtin_amdgcn_wmma_f32_16x16x32_bf16(false, a_hi[t], false, bl[t], (short)0, acc, false, false);
        }

        // dist (minus the row-constant ||z_e||^2) = ||cb||^2 - 2*cross
        const int kidx = kt * TILE_K + l16;
        #pragma unroll
        for (int j = 0; j < 8; ++j) {
            float d = cbsq - 2.0f * acc[j];
            if (d < best[j]) { best[j] = d; bidx[j] = kidx; }   // strict < keeps lowest idx
        }

        __syncthreads();  // next tile fully staged; previous buffer free for reuse
    }

    // ---- Per-row argmin across the 16 lanes of each half, then gather + loss.
    // C/D layout: slot j, lanes<16 -> row n0+j; lanes>=16 -> row n0+j+8.
    float lsum = 0.f;
    #pragma unroll
    for (int j = 0; j < 8; ++j) {
        float v = best[j];
        int   ix = bidx[j];
        #pragma unroll
        for (int m = 1; m < 16; m <<= 1) {
            float ov = __shfl_xor(v, m, 32);
            int   oi = __shfl_xor(ix, m, 32);
            if (ov < v || (ov == v && oi < ix)) { v = ov; ix = oi; }
        }
        const int row = n0 + j + 8 * half;
        const float* cq = cb  + (size_t)ix  * C_DIM + l16 * 8;
        const float* zr = z_e + (size_t)row * C_DIM + l16 * 8;
        float4 c0 = *(const float4*)(cq);
        float4 c1 = *(const float4*)(cq + 4);
        float4 z0 = *(const float4*)(zr);
        float4 z1 = *(const float4*)(zr + 4);
        *(float4*)(out_zq + (size_t)row * C_DIM + l16 * 8)     = c0;   // z_q_st == z_q
        *(float4*)(out_zq + (size_t)row * C_DIM + l16 * 8 + 4) = c1;
        float dx;
        dx = z0.x - c0.x; lsum += dx * dx;
        dx = z0.y - c0.y; lsum += dx * dx;
        dx = z0.z - c0.z; lsum += dx * dx;
        dx = z0.w - c0.w; lsum += dx * dx;
        dx = z1.x - c1.x; lsum += dx * dx;
        dx = z1.y - c1.y; lsum += dx * dx;
        dx = z1.z - c1.z; lsum += dx * dx;
        dx = z1.w - c1.w; lsum += dx * dx;
        if (l16 == 0) out_idx[row] = (float)ix;
    }

    // ---- deterministic loss partial per block
    #pragma unroll
    for (int m = 1; m < 32; m <<= 1) lsum += __shfl_xor(lsum, m, 32);
    if (lane == 0) s_loss[wave] = lsum;
    __syncthreads();
    if (tid == 0) {
        float s = 0.f;
        #pragma unroll
        for (int w = 0; w < WAVES_PER_BLOCK; ++w) s += s_loss[w];
        partial[blockIdx.x] = s;
    }
}

// ---------------------------------------------------------------------------
// Kernel 2: deterministic fixed-order reduction of block partials -> loss.
// ---------------------------------------------------------------------------
__global__ __launch_bounds__(256) void vq_loss_finalize(
    const float* __restrict__ partial, int nparts,
    float* __restrict__ out_loss, float scale)
{
    __shared__ float s[256];
    const int t = threadIdx.x;
    float v = 0.f;
    for (int i = t; i < nparts; i += 256) v += partial[i];
    s[t] = v;
    __syncthreads();
    for (int off = 128; off > 0; off >>= 1) {
        if (t < off) s[t] += s[t + off];
        __syncthreads();
    }
    if (t == 0) *out_loss = s[0] * scale;
}

extern "C" void kernel_launch(void* const* d_in, const int* in_sizes, int n_in,
                              void* d_out, int out_size, void* d_ws, size_t ws_size,
                              hipStream_t stream)
{
    const float* z_e = (const float*)d_in[0];   // [N, 128] fp32
    const float* cb  = (const float*)d_in[1];   // [K, 128] fp32
    const int N = in_sizes[0] / C_DIM;          // 65536
    const int K = in_sizes[1] / C_DIM;          // 1024

    float* out      = (float*)d_out;
    float* out_zq   = out;                       // [N*128]
    float* out_loss = out + (size_t)N * C_DIM;   // [1]
    float* out_idx  = out_loss + 1;              // [N] (indices as fp32 values)

    float* partial  = (float*)d_ws;              // [blocks]  (<= 1024 floats)
    float* cbsq_g   = (float*)d_ws + 1024;       // [K]

    vq_cbsq_kernel<<<(K + 255) / 256, 256, 0, stream>>>(cb, cbsq_g, K);

    const int blocks = N / ROWS_PER_BLOCK;       // 512
    vq_argmin_kernel<<<blocks, 256, 0, stream>>>(z_e, cb, cbsq_g,
                                                 out_zq, out_idx, partial, K);

    const float scale = 1.75f / ((float)N * (float)C_DIM);  // 0.75*q + e = 1.75*mse
    vq_loss_finalize<<<1, 256, 0, stream>>>(partial, blocks, out_loss, scale);
}
